// Net_57105885168298
// MI455X (gfx1250) — matrix operand order repaired and checked by reading, not compiled, not run on hardware
//
#include <hip/hip_runtime.h>
#include <hip/hip_bf16.h>

typedef __attribute__((ext_vector_type(16))) _Float16 v16h;
typedef __attribute__((ext_vector_type(8)))  float    v8f;
typedef unsigned int u32;
typedef __attribute__((ext_vector_type(4))) unsigned int u32x4;
typedef __attribute__((ext_vector_type(8))) int          i32x8;
typedef __attribute__((ext_vector_type(4))) int          i32x4;

#if defined(__has_builtin)
#if __has_builtin(__builtin_amdgcn_tensor_load_to_lds) && __has_builtin(__builtin_amdgcn_s_wait_tensorcnt)
#define USE_TDM 1
#endif
#endif

// ---------------- problem constants ----------------
constexpr int NB   = 16;        // graphs
constexpr int NPG  = 2048;      // nodes per graph
constexpr int NN   = NB * NPG;  // 32768 nodes
constexpr int CIN  = 64;
constexpr int HID  = 128;
constexpr int KCL  = 32;        // clusters
constexpr int EE   = 524288;    // edges

// ---------------- workspace layout (bytes) ----------------
constexpr size_t O_SSUM = 0;                       // 1024 f32
constexpr size_t O_SSQ  = O_SSUM + 4096;           // 1024 f32
constexpr size_t O_CA   = O_SSQ  + 4096;           // 512 f32
constexpr size_t O_CSZ  = O_CA   + 2048;           // 512 f32
constexpr size_t O_TRA  = O_CSZ  + 2048;           // 16 f32
constexpr size_t ZB_BYTES = O_TRA + 64;            // 12352 — zeroed each call
constexpr size_t O_XN16 = 12800;                   // N*CIN f16    (4 MB)
constexpr size_t O_W1PK = O_XN16 + 4194304;        // 32*128 u32
constexpr size_t O_WPPK = O_W1PK + 16384;          // 64*32 u32
constexpr size_t O_H1   = O_WPPK + 8192;           // N*HID f32    (16 MB)
constexpr size_t O_DEG  = O_H1   + 16777216;       // N f32
constexpr size_t O_ODEG = O_DEG  + 131072;         // N f32
constexpr size_t O_HAGG = O_ODEG + 131072;         // N*HID f32    (16 MB, zeroed)
constexpr size_t O_H16  = O_HAGG + 16777216;       // N*HID f16    (8 MB)
constexpr size_t O_HPK  = O_H16  + 8388608;        // (N/2)*HID u32 (8 MB)
constexpr size_t O_LOG  = O_HPK  + 8388608;        // N*K f32     (4 MB)
constexpr size_t O_SPK  = O_LOG  + 4194304;        // (N/2)*K u32 (2 MB)
constexpr size_t O_OXP  = O_SPK  + 2097152;        // B*K*HID f32 (256 KB, zeroed)
constexpr size_t O_SS   = O_OXP  + 262144;         // B*K*K f32   (64 KB)

// ---------------- helpers ----------------
union Frag { v16h h; u32 u[8]; uint4 q[2]; };

__device__ inline v8f wmma16(const Frag& a, const Frag& b, v8f c) {
  return __builtin_amdgcn_wmma_f32_16x16x32_f16(false, a.h, false, b.h, (short)0, c, false, false);
}
__device__ inline u32 pkh(float a, float b) {
  union { _Float16 h[2]; u32 u; } z;
  z.h[0] = (_Float16)a; z.h[1] = (_Float16)b;
  return z.u;
}
__device__ inline float blockRed256(float v, float* red) {
  int t = threadIdx.x;
  red[t] = v; __syncthreads();
  for (int o = 128; o > 0; o >>= 1) { if (t < o) red[t] += red[t + o]; __syncthreads(); }
  float r = red[0]; __syncthreads();
  return r;
}

#ifdef USE_TDM
// Issue a 1D TDM tile copy: ndw dwords from global -> LDS (D# per ISA 08 §8.3/8.4).
// 6-arg builtin form (clang-23 / therock-10.0 headers): extra descriptor group zeroed.
__device__ inline void tdm_load_1d(u32 lds_off, const void* gaddr, u32 ndw) {
  unsigned long long ga = (unsigned long long)gaddr;
  u32x4 g0;
  g0[0] = 1u;                                          // count=1, user descriptor
  g0[1] = lds_off;                                     // lds_addr [63:32]
  g0[2] = (u32)ga;                                     // global_addr [95:64]
  g0[3] = (u32)((ga >> 32) & 0x1FFFFFFu) | (2u << 30); // addr[56:32] | type=2
  i32x8 g1;
  g1[0] = (int)(2u << 16);                 // workgroup_mask=0, data_size=4B
  g1[1] = (int)((ndw & 0xFFFFu) << 16);    // tensor_dim0 [79:48] lo
  g1[2] = (int)((ndw >> 16) | (1u << 16)); // tensor_dim0 hi | tensor_dim1=1 lo
  g1[3] = (int)((ndw & 0xFFFFu) << 16);    // tensor_dim1 hi=0 | tile_dim0 [127:112]
  g1[4] = 1;                               // tile_dim1=1, tile_dim2=0
  g1[5] = (int)ndw;                        // tensor_dim0_stride lo32
  g1[6] = 0;
  g1[7] = 0;
  i32x4 gz4 = {0, 0, 0, 0};
  i32x8 gz8 = {0, 0, 0, 0, 0, 0, 0, 0};
  __builtin_amdgcn_tensor_load_to_lds(g0, g1, gz4, gz4, gz8, 0);
}
#endif

// ---------------- 1. GraphNorm statistics (single pass) ----------------
__global__ __launch_bounds__(256) void k_gn_stats(const float* __restrict__ x,
                                                  float* __restrict__ ssum,
                                                  float* __restrict__ ssq) {
  int nodebase = blockIdx.x * 128;           // 256 blocks cover 32768 nodes
  int c  = threadIdx.x & 63;
  int n0 = threadIdx.x >> 6;
  float s = 0.f, q = 0.f;
  for (int n = n0; n < 128; n += 4) {
    float v = x[(size_t)(nodebase + n) * CIN + c];
    s += v; q += v * v;
  }
  int b = nodebase >> 11;
  atomicAdd(&ssum[b * CIN + c], s);
  atomicAdd(&ssq [b * CIN + c], q);
}

// ---------------- 2. normalize -> f16 (packed pairs along feature) ----------------
__global__ __launch_bounds__(256) void k_gn_norm(const float* __restrict__ x,
                                                 const float* __restrict__ ssum,
                                                 const float* __restrict__ ssq,
                                                 const float* __restrict__ w,
                                                 const float* __restrict__ bias,
                                                 const float* __restrict__ msc,
                                                 u32* __restrict__ xn16) {
  int p = blockIdx.x * 256 + threadIdx.x;    // pair index over N*CIN/2
  int n  = p >> 5;
  int c0 = (p & 31) * 2;
  int b  = n >> 11;
  const float inv_cnt = 1.f / 2048.f;
  float r[2];
#pragma unroll
  for (int j = 0; j < 2; ++j) {
    int c = c0 + j;
    float mu  = ssum[b * CIN + c] * inv_cnt;
    float a   = msc[c];
    float var = ssq[b * CIN + c] * inv_cnt - mu * mu * (2.f * a - a * a);
    float xv  = x[(size_t)n * CIN + c];
    float o   = xv - mu * a;
    r[j] = w[c] * o * rsqrtf(var + 1e-5f) + bias[c];
  }
  xn16[p] = pkh(r[0], r[1]);
}

// ---------------- 3. pre-pack weights into WMMA-B pair layout ----------------
__global__ __launch_bounds__(256) void k_prep_w(const float* __restrict__ W1,
                                                const float* __restrict__ Wp,
                                                u32* __restrict__ w1pk,
                                                u32* __restrict__ wppk) {
  int i = blockIdx.x * 256 + threadIdx.x;
  if (i < 32 * 128) {                        // W1[64][128] -> pairs along K
    int k2 = i >> 7, nn = i & 127;
    w1pk[i] = pkh(W1[(2 * k2) * HID + nn], W1[(2 * k2 + 1) * HID + nn]);
  } else {
    int j = i - 32 * 128;
    if (j < 64 * 32) {                       // Wp[128][32] -> pairs along K
      int k2 = j >> 5, nn = j & 31;
      wppk[j] = pkh(Wp[(2 * k2) * KCL + nn], Wp[(2 * k2 + 1) * KCL + nn]);
    }
  }
}

// ---------------- 4. degrees ----------------
__global__ __launch_bounds__(256) void k_deg_init(float* __restrict__ deg,
                                                  float* __restrict__ odeg) {
  int n = blockIdx.x * 256 + threadIdx.x;
  if (n < NN) { deg[n] = 1.f; odeg[n] = 0.f; }   // self-loop gives in-degree >= 1
}
__global__ __launch_bounds__(256) void k_deg_edges(const int* __restrict__ ei,
                                                   float* __restrict__ deg,
                                                   float* __restrict__ odeg) {
  int e = blockIdx.x * 256 + threadIdx.x;
  if (e < EE) {
    int src = ei[e], dst = ei[EE + e];
    atomicAdd(&deg[dst], 1.f);
    atomicAdd(&odeg[src], 1.f);
  }
}

// ---------------- 5. GEMM1: h1 = xn @ W1   [N,64]x[64,128] (WMMA f16) ----------------
__global__ __launch_bounds__(256) void k_gemm1(const _Float16* __restrict__ xn16,
                                               const u32* __restrict__ w1pk,
                                               float* __restrict__ h1) {
  int wave = threadIdx.x >> 5, lane = threadIdx.x & 31;
  int hi = lane >> 4;
  int mbase = blockIdx.x * 16;
  int row = mbase + (lane & 15);
  int col = wave * 16 + (lane & 15);
  v8f acc = {};
#pragma unroll
  for (int step = 0; step < 2; ++step) {
    Frag a, b;
    const uint4* ap = (const uint4*)(xn16 + (size_t)row * CIN + step * 32 + hi * 8);
    a.q[0] = ap[0];                          // K pairs base..base+3
    a.q[1] = ap[2];                          // K pairs base+8..base+11
    int kc2 = step * 16 + hi * 4;
#pragma unroll
    for (int j = 0; j < 8; ++j) b.u[j] = w1pk[(kc2 + j + (j & 4)) * HID + col];
    acc = wmma16(a, b, acc);
  }
  int rbase = mbase + hi * 8;
#pragma unroll
  for (int r = 0; r < 8; ++r) h1[(size_t)(rbase + r) * HID + col] = acc[r];
}

// ---------------- 6. sparse aggregation: h_agg[dst] += norm * h1[src] ----------------
__global__ __launch_bounds__(256) void k_agg(const int* __restrict__ ei,
                                             const float* __restrict__ deg,
                                             const float* __restrict__ h1,
                                             float* __restrict__ hagg) {
  int e = blockIdx.x * 8 + (threadIdx.x >> 5);
  int lane = threadIdx.x & 31;
  int src = ei[e], dst = ei[EE + e];
  float nrm = rsqrtf(deg[src]) * rsqrtf(deg[dst]);
  float4 v = ((const float4*)(h1 + (size_t)src * HID))[lane];
  float* o = hagg + (size_t)dst * HID + lane * 4;
  atomicAdd(o + 0, nrm * v.x);
  atomicAdd(o + 1, nrm * v.y);
  atomicAdd(o + 2, nrm * v.z);
  atomicAdd(o + 3, nrm * v.w);
}

// ---------------- 7. epilogue: h = relu(hagg + h1/deg + b1) -> f16 ----------------
__global__ __launch_bounds__(256) void k_epi1(const float* __restrict__ hagg,
                                              const float* __restrict__ h1,
                                              const float* __restrict__ deg,
                                              const float* __restrict__ b1,
                                              u32* __restrict__ h16) {
  int i = blockIdx.x * 256 + threadIdx.x;    // over N*HID/4
  int n = i >> 5;
  int f = (i & 31) * 4;
  float inv = 1.f / deg[n];                  // self-loop norm = dis[n]^2
  float4 ha = *(const float4*)(hagg + (size_t)n * HID + f);
  float4 hv = *(const float4*)(h1   + (size_t)n * HID + f);
  float4 bb = *(const float4*)(b1 + f);
  float r0 = fmaxf(0.f, ha.x + hv.x * inv + bb.x);
  float r1 = fmaxf(0.f, ha.y + hv.y * inv + bb.y);
  float r2 = fmaxf(0.f, ha.z + hv.z * inv + bb.z);
  float r3 = fmaxf(0.f, ha.w + hv.w * inv + bb.w);
  size_t p = ((size_t)n * HID + f) >> 1;
  h16[p]     = pkh(r0, r1);
  h16[p + 1] = pkh(r2, r3);
}

// node-pair-packed copy of h (WMMA-B layout for GEMM3)
__global__ __launch_bounds__(256) void k_pack_h(const unsigned short* __restrict__ h16u,
                                                u32* __restrict__ hpk) {
  int i = blockIdx.x * 256 + threadIdx.x;    // over (N/2)*HID
  int p = i >> 7, f = i & 127;
  u32 lo = h16u[(size_t)(2 * p) * HID + f];
  u32 hh = h16u[(size_t)(2 * p + 1) * HID + f];
  hpk[i] = lo | (hh << 16);
}

// ---------------- 8. GEMM2: logits = h @ Wp + bp  [N,128]x[128,32] (WMMA) ----------------
__global__ __launch_bounds__(256) void k_gemm2(const _Float16* __restrict__ h16,
                                               const u32* __restrict__ wppk,
                                               const float* __restrict__ bp,
                                               float* __restrict__ logits) {
  int wave = threadIdx.x >> 5, lane = threadIdx.x & 31;
  int hi = lane >> 4;
  int mbase = blockIdx.x * 64 + (wave >> 1) * 16;
  int row = mbase + (lane & 15);
  int col = (wave & 1) * 16 + (lane & 15);
  v8f acc = {};
#pragma unroll
  for (int step = 0; step < 4; ++step) {
    Frag a, b;
    const uint4* ap = (const uint4*)(h16 + (size_t)row * HID + step * 32 + hi * 8);
    a.q[0] = ap[0];
    a.q[1] = ap[2];
    int kc2 = step * 16 + hi * 4;
#pragma unroll
    for (int j = 0; j < 8; ++j) b.u[j] = wppk[(kc2 + j + (j & 4)) * KCL + col];
    acc = wmma16(a, b, acc);
  }
  int rbase = mbase + hi * 8;
  float bc = bp[col];
#pragma unroll
  for (int r = 0; r < 8; ++r) logits[(size_t)(rbase + r) * KCL + col] = acc[r] + bc;
}

// ---------------- 9. softmax rows -> s (d_out) + f16 packed + ca/csize ----------------
__global__ __launch_bounds__(256) void k_softmax(const float* __restrict__ logits,
                                                 float* __restrict__ s_out,
                                                 u32* __restrict__ spk,
                                                 float* __restrict__ csize,
                                                 float* __restrict__ ca,
                                                 const float* __restrict__ odeg) {
  __shared__ float sm[8][KCL];
  int wave = threadIdx.x >> 5, lane = threadIdx.x & 31;
  int row = blockIdx.x * 8 + wave;
  float v = logits[(size_t)row * KCL + lane];
  float mx = v;
  for (int o = 16; o; o >>= 1) mx = fmaxf(mx, __shfl_xor(mx, o, 32));
  float e = __expf(v - mx);
  float sum = e;
  for (int o = 16; o; o >>= 1) sum += __shfl_xor(sum, o, 32);
  float sv = e / sum;
  s_out[(size_t)row * KCL + lane] = sv;
  sm[wave][lane] = sv;
  int b = row >> 11;
  atomicAdd(&csize[b * KCL + lane], sv);
  atomicAdd(&ca[b * KCL + lane], sv * odeg[row]);
  __syncthreads();
  if (threadIdx.x < 128) {                   // pack node pairs for GEMM3 A
    int p = threadIdx.x >> 5, m = threadIdx.x & 31;
    spk[(blockIdx.x * 4 + p) * KCL + m] = pkh(sm[2 * p][m], sm[2 * p + 1][m]);
  }
}

// ---------------- 10. GEMM3: out_x_pre[b] = s^T @ h per graph ----------------
// TDM stages the 128-node chunk (packed pair-rows) into LDS; WMMA reads via ds.
__global__ __launch_bounds__(256) void k_gemm3(const u32* __restrict__ spk,
                                               const u32* __restrict__ hpk,
                                               float* __restrict__ oxp) {
  __shared__ __align__(16) u32 ldsH[64 * HID];   // 32 KB: hpk rows kc2..kc2+63
  __shared__ __align__(16) u32 ldsS[64 * KCL];   //  8 KB: spk rows kc2..kc2+63
  int wave = threadIdx.x >> 5, lane = threadIdx.x & 31;
  int hi = lane >> 4;
  int g = blockIdx.y;
  int gn0 = g * NPG + blockIdx.x * 128;          // 16 K-segments of 128 nodes
  int kc2base = gn0 >> 1;                        // 64 pair-rows per chunk
  int col = wave * 16 + (lane & 15);             // hid
  int m0 = lane & 15;

#ifdef USE_TDM
  if (threadIdx.x == 0) {                        // wave 0 issues both DMA tiles
    tdm_load_1d((u32)(size_t)(void*)ldsH, hpk + (size_t)kc2base * HID, 64 * HID);
    tdm_load_1d((u32)(size_t)(void*)ldsS, spk + (size_t)kc2base * KCL, 64 * KCL);
    __builtin_amdgcn_s_wait_tensorcnt(0);
  }
#else
  for (int i = threadIdx.x; i < 64 * HID; i += 256) ldsH[i] = hpk[(size_t)kc2base * HID + i];
  for (int i = threadIdx.x; i < 64 * KCL; i += 256) ldsS[i] = spk[(size_t)kc2base * KCL + i];
#endif
  __syncthreads();

  v8f acc0 = {}, acc1 = {};
#pragma unroll
  for (int kk = 0; kk < 128; kk += 32) {
    Frag a0, a1, b;
    int lc2 = kk >> 1;                           // local pair-row base
    int off = hi * 4;
#pragma unroll
    for (int j = 0; j < 4; ++j) {
      a0.u[j]     = ldsS[(lc2 + off + j)     * KCL + m0];
      a0.u[4 + j] = ldsS[(lc2 + off + 8 + j) * KCL + m0];
      a1.u[j]     = ldsS[(lc2 + off + j)     * KCL + 16 + m0];
      a1.u[4 + j] = ldsS[(lc2 + off + 8 + j) * KCL + 16 + m0];
    }
#pragma unroll
    for (int j = 0; j < 8; ++j) b.u[j] = ldsH[(lc2 + hi * 4 + j + (j & 4)) * HID + col];
    acc0 = wmma16(a0, b, acc0);
    acc1 = wmma16(a1, b, acc1);
  }
  float* o = oxp + (size_t)g * KCL * HID;
#pragma unroll
  for (int r = 0; r < 8; ++r) {
    atomicAdd(&o[(r + hi * 8) * HID + col], acc0[r]);
    atomicAdd(&o[(16 + r + hi * 8) * HID + col], acc1[r]);
  }
}

// ---------------- 11. selu + log_softmax over HID -> d_out ----------------
__global__ __launch_bounds__(128) void k_oxfinal(const float* __restrict__ oxp,
                                                 float* __restrict__ out_x) {
  __shared__ float red[128];
  int row = blockIdx.x, t = threadIdx.x;
  float v = oxp[(size_t)row * HID + t];
  const float sc = 1.0507009873554805f, al = 1.6732632423543772f;
  v = v > 0.f ? sc * v : sc * al * (__expf(v) - 1.f);
  red[t] = v; __syncthreads();
  for (int o = 64; o; o >>= 1) { if (t < o) red[t] = fmaxf(red[t], red[t + o]); __syncthreads(); }
  float mx = red[0]; __syncthreads();
  red[t] = __expf(v - mx); __syncthreads();
  for (int o = 64; o; o >>= 1) { if (t < o) red[t] += red[t + o]; __syncthreads(); }
  float lse = logf(red[0]);
  out_x[(size_t)row * HID + t] = v - mx - lse;
}

// ---------------- 12. ss = s^T s per graph ----------------
__global__ __launch_bounds__(256) void k_ss(const float* __restrict__ s,
                                            float* __restrict__ ssm) {
  __shared__ float sl[64 * KCL];
  int g = blockIdx.x;
  float acc[4] = {0.f, 0.f, 0.f, 0.f};
  int p0 = threadIdx.x * 4;
  for (int c = 0; c < NPG; c += 64) {
    __syncthreads();
    for (int i = threadIdx.x; i < 64 * KCL; i += 256)
      sl[i] = s[(size_t)(g * NPG + c + (i >> 5)) * KCL + (i & 31)];
    __syncthreads();
    for (int n = 0; n < 64; ++n) {
      const float* r = &sl[n * KCL];
#pragma unroll
      for (int j = 0; j < 4; ++j) {
        int p = p0 + j;
        acc[j] += r[p >> 5] * r[p & 31];
      }
    }
  }
#pragma unroll
  for (int j = 0; j < 4; ++j) ssm[g * KCL * KCL + p0 + j] = acc[j];
}

// ---------------- 13. trace(out_adj) = sum_edges dot(s[src], s[dst]) ----------------
__global__ __launch_bounds__(256) void k_trace(const int* __restrict__ ei,
                                               const float* __restrict__ s,
                                               float* __restrict__ traceA) {
  int e = blockIdx.x * 8 + (threadIdx.x >> 5);
  int lane = threadIdx.x & 31;
  int src = ei[e], dst = ei[EE + e];
  float p = s[(size_t)src * KCL + lane] * s[(size_t)dst * KCL + lane];
  for (int o = 16; o; o >>= 1) p += __shfl_xor(p, o, 32);
  if (lane == 0) atomicAdd(&traceA[src >> 11], p);
}

// ---------------- 14. final scalar loss ----------------
__global__ __launch_bounds__(256) void k_loss(const float* __restrict__ odeg,
                                              const float* __restrict__ ca,
                                              const float* __restrict__ csize,
                                              const float* __restrict__ ssm,
                                              const float* __restrict__ traceA,
                                              float* __restrict__ loss_out) {
  __shared__ float red[256];
  int t = threadIdx.x;
  float sspec = 0.f, sorth = 0.f, sclus = 0.f;   // meaningful on t==0 only
  const float isq = rsqrtf(32.f);
  for (int b = 0; b < NB; ++b) {
    float p = 0.f;
    for (int i = t; i < NPG; i += 256) p += odeg[b * NPG + i];
    float m = blockRed256(p, red) * 0.5f;
    p = 0.f;
    for (int i = t; i < KCL * KCL; i += 256) { float v = ssm[b * KCL * KCL + i]; p += v * v; }
    float ssn = sqrtf(blockRed256(p, red));
    p = 0.f;
    for (int i = t; i < KCL * KCL; i += 256) {
      float v = ssm[b * KCL * KCL + i] / ssn;
      if ((i >> 5) == (i & 31)) v -= isq;
      p += v * v;
    }
    float orth = sqrtf(blockRed256(p, red));
    if (t == 0) {
      float catr = 0.f, cn = 0.f;
      for (int k = 0; k < KCL; ++k) {
        float c = ca[b * KCL + k];    catr += c * c;
        float d = csize[b * KCL + k]; cn   += d * d;
      }
      sspec += -(traceA[b] - catr / (2.f * m)) / (2.f * m);
      sorth += orth;
      sclus += sqrtf(cn) / (float)NPG * sqrtf(32.f) - 1.f;
    }
    __syncthreads();
  }
  if (t == 0) loss_out[0] = (sspec + sorth + sclus) / (float)NB;
}

// ---------------- host launcher ----------------
extern "C" void kernel_launch(void* const* d_in, const int* in_sizes, int n_in,
                              void* d_out, int out_size, void* d_ws, size_t ws_size,
                              hipStream_t stream) {
  (void)in_sizes; (void)n_in; (void)out_size; (void)ws_size;
  const float* x    = (const float*)d_in[0];
  const float* gw   = (const float*)d_in[1];
  const float* gb   = (const float*)d_in[2];
  const float* gms  = (const float*)d_in[3];
  const float* W1   = (const float*)d_in[4];
  const float* b1   = (const float*)d_in[5];
  const float* Wp   = (const float*)d_in[6];
  const float* bp   = (const float*)d_in[7];
  const int*   ei   = (const int*)d_in[8];

  char* ws = (char*)d_ws;
  float* ssum   = (float*)(ws + O_SSUM);
  float* ssq    = (float*)(ws + O_SSQ);
  float* ca     = (float*)(ws + O_CA);
  float* csize  = (float*)(ws + O_CSZ);
  float* traceA = (float*)(ws + O_TRA);
  _Float16* xn16 = (_Float16*)(ws + O_XN16);
  u32*  w1pk  = (u32*)(ws + O_W1PK);
  u32*  wppk  = (u32*)(ws + O_WPPK);
  float* h1   = (float*)(ws + O_H1);
  float* deg  = (float*)(ws + O_DEG);
  float* odeg = (float*)(ws + O_ODEG);
  float* hagg = (float*)(ws + O_HAGG);
  _Float16* h16 = (_Float16*)(ws + O_H16);
  u32*  hpk   = (u32*)(ws + O_HPK);
  float* logits = (float*)(ws + O_LOG);
  u32*  spk   = (u32*)(ws + O_SPK);
  float* oxp  = (float*)(ws + O_OXP);
  float* ssm  = (float*)(ws + O_SS);

  float* out_x  = (float*)d_out;                       // [B,K,HID] = 65536
  float* out_l  = (float*)d_out + NB * KCL * HID;      // scalar loss
  float* s_out  = out_l + 1;                           // [B,NPG,K]

  (void)hipMemsetAsync(ws + O_SSUM, 0, ZB_BYTES, stream);    // stats/ca/csize/trace
  (void)hipMemsetAsync(ws + O_HAGG, 0, (size_t)NN * HID * 4, stream);
  (void)hipMemsetAsync(ws + O_OXP,  0, (size_t)NB * KCL * HID * 4, stream);

  k_gn_stats<<<NN / 128, 256, 0, stream>>>(x, ssum, ssq);
  k_gn_norm <<<NN * CIN / 512, 256, 0, stream>>>(x, ssum, ssq, gw, gb, gms, (u32*)xn16);
  k_prep_w  <<<24, 256, 0, stream>>>(W1, Wp, w1pk, wppk);
  k_deg_init<<<NN / 256, 256, 0, stream>>>(deg, odeg);
  k_deg_edges<<<EE / 256, 256, 0, stream>>>(ei, deg, odeg);
  k_gemm1   <<<NN / 16, 256, 0, stream>>>(xn16, w1pk, h1);
  k_agg     <<<EE / 8, 256, 0, stream>>>(ei, deg, h1, hagg);
  k_epi1    <<<NN * HID / 1024, 256, 0, stream>>>(hagg, h1, deg, b1, (u32*)h16);
  k_pack_h  <<<(NN / 2) * HID / 256, 256, 0, stream>>>((const unsigned short*)h16, hpk);
  k_gemm2   <<<NN / 64, 256, 0, stream>>>(h16, wppk, bp, logits);
  k_softmax <<<NN / 8, 256, 0, stream>>>(logits, s_out, spk, csize, ca, odeg);
  k_gemm3   <<<dim3(16, NB), 256, 0, stream>>>(spk, hpk, oxp);
  k_oxfinal <<<NB * KCL, 128, 0, stream>>>(oxp, out_x);
  k_ss      <<<NB, 256, 0, stream>>>(s_out, ssm);
  k_trace   <<<EE / 8, 256, 0, stream>>>(ei, s_out, traceA);
  k_loss    <<<1, 256, 0, stream>>>(odeg, ca, csize, ssm, traceA, out_l);
}
